// Model_37684043055863
// MI455X (gfx1250) — compile-verified, matrix operand order checked by reading, not run on hardware
//
#include <hip/hip_runtime.h>

// ---------------------------------------------------------------------------
// PCN forward for MI455X (gfx1250): bf16 WMMA GEMMs + algebraic fusion of the
// broadcast-heavy layers.  All activations stored point-major [P, C] (C
// contiguous) so WMMA A and B fragments are two 16-byte contiguous loads/lane.
// Heavy GEMMs use a 64x64 wave tile (16 v_wmma per K-step, 32 FLOP/B fetched);
// skinny GEMMs (P=32 or M=32) use a 32x32 wave tile.
// ---------------------------------------------------------------------------

typedef __bf16 bf16_t;
typedef __attribute__((ext_vector_type(16))) __bf16 v16bf;
typedef __attribute__((ext_vector_type(8)))  float  v8f;

union FragU { v16bf v; uint4 q[2]; };
union PackU { bf16_t h[8]; uint4 q; };

// Per-lane fragment load for 16-bit WMMA operands (ISA 7.12.2):
// lane half g=0 holds K = {0..7, 16..23}, g=1 holds K = {8..15, 24..31}.
__device__ __forceinline__ v16bf ldfrag(const bf16_t* row, int g) {
  FragU f;
  f.q[0] = *(const uint4*)(row + g * 8);
  f.q[1] = *(const uint4*)(row + 16 + g * 8);
  return f.v;
}

__device__ __forceinline__ v8f wmma_bf16(v16bf a, v16bf b, v8f c) {
  return __builtin_amdgcn_wmma_f32_16x16x32_bf16(false, a, false, b,
                                                 (short)0, c, false, false);
}

// ---------------------------------------------------------------------------
// Heavy GEMM:  Y[p, m] = act( sum_k W[m,k] * X[p,k] + bias[m] + cbias[p/cdiv, m] )
// Block = 8 waves = 2(M) x 4(P); wave tile 64x64 = 4x4 WMMA tiles.
// Requires M % 64 == 0 (per-wave), P % 256 handled by guard at 64 granularity.
// ---------------------------------------------------------------------------
__global__ __launch_bounds__(256)
void k_gemm_big(const bf16_t* __restrict__ W, int ldw,
                const bf16_t* __restrict__ X, int ldx,
                const float* __restrict__ bias,
                const float* __restrict__ cbias, int cdiv,
                bf16_t* __restrict__ Y,
                int M, int K, int P, int relu)
{
  const int wave  = threadIdx.x >> 5;
  const int lane  = threadIdx.x & 31;
  const int mBase = blockIdx.y * 128 + (wave >> 2) * 64;
  const int pBase = blockIdx.x * 256 + (wave & 3) * 64;
  if (mBase >= M || pBase >= P) return;
  const int g = lane >> 4;
  const int s = lane & 15;

  v8f acc[4][4];
#pragma unroll
  for (int i = 0; i < 4; ++i)
#pragma unroll
    for (int j = 0; j < 4; ++j) acc[i][j] = (v8f)0.0f;

  const bf16_t* wr[4];
  const bf16_t* xr[4];
#pragma unroll
  for (int t = 0; t < 4; ++t) {
    wr[t] = W + (size_t)(mBase + 16 * t + s) * ldw;
    xr[t] = X + (size_t)(pBase + 16 * t + s) * ldx;
  }

  for (int k0 = 0; k0 < K; k0 += 32) {
    v16bf a[4], b[4];
#pragma unroll
    for (int t = 0; t < 4; ++t) a[t] = ldfrag(wr[t] + k0, g);
#pragma unroll
    for (int t = 0; t < 4; ++t) b[t] = ldfrag(xr[t] + k0, g);
#pragma unroll
    for (int im = 0; im < 4; ++im)
#pragma unroll
      for (int ip = 0; ip < 4; ++ip)
        acc[im][ip] = wmma_bf16(a[im], b[ip], acc[im][ip]);
  }

  // D layout (ISA): VGPR r, lane -> row m = 8*g + r, col n = s.
#pragma unroll
  for (int im = 0; im < 4; ++im) {
#pragma unroll
    for (int ip = 0; ip < 4; ++ip) {
      const int mrow = mBase + im * 16 + g * 8;
      const int pcol = pBase + ip * 16 + s;
      v8f a = acc[im][ip];
      if (bias) {
#pragma unroll
        for (int r = 0; r < 8; ++r) a[r] += bias[mrow + r];
      }
      if (cbias) {
        const float* cb = cbias + (size_t)(pcol / cdiv) * M + mrow;
#pragma unroll
        for (int r = 0; r < 8; ++r) a[r] += cb[r];
      }
      if (relu) {
#pragma unroll
        for (int r = 0; r < 8; ++r) a[r] = fmaxf(a[r], 0.0f);
      }
      PackU o;
#pragma unroll
      for (int r = 0; r < 8; ++r) o.h[r] = (bf16_t)a[r];
      *(uint4*)(Y + (size_t)pcol * M + mrow) = o.q;
    }
  }
}

// ---------------------------------------------------------------------------
// Skinny GEMM (decoder P=32, T_global, M=32 folding output).
// Block = 8 waves = 4(M) x 2(P); wave tile 32x32 = 2x2 WMMA tiles.
// ---------------------------------------------------------------------------
__global__ __launch_bounds__(256)
void k_gemm(const bf16_t* __restrict__ W, int ldw,
            const bf16_t* __restrict__ X, int ldx,
            const float* __restrict__ bias,
            const float* __restrict__ cbias, int cdiv,
            bf16_t* __restrict__ Y, float* __restrict__ Yf,
            int M, int K, int P, int relu)
{
  const int wave  = threadIdx.x >> 5;
  const int lane  = threadIdx.x & 31;
  const int mBase = blockIdx.y * 128 + (wave >> 1) * 32;
  const int pBase = blockIdx.x * 64  + (wave & 1) * 32;
  if (mBase >= M || pBase >= P) return;
  const int g = lane >> 4;
  const int s = lane & 15;

  v8f acc[2][2];
#pragma unroll
  for (int i = 0; i < 2; ++i)
#pragma unroll
    for (int j = 0; j < 2; ++j) acc[i][j] = (v8f)0.0f;

  const bf16_t* w0 = W + (size_t)(mBase + s)      * ldw;
  const bf16_t* w1 = W + (size_t)(mBase + 16 + s) * ldw;
  const bf16_t* x0 = X + (size_t)(pBase + s)      * ldx;
  const bf16_t* x1 = X + (size_t)(pBase + 16 + s) * ldx;

  for (int k0 = 0; k0 < K; k0 += 32) {
    v16bf a0 = ldfrag(w0 + k0, g);
    v16bf a1 = ldfrag(w1 + k0, g);
    v16bf b0 = ldfrag(x0 + k0, g);
    v16bf b1 = ldfrag(x1 + k0, g);
    acc[0][0] = wmma_bf16(a0, b0, acc[0][0]);
    acc[0][1] = wmma_bf16(a0, b1, acc[0][1]);
    acc[1][0] = wmma_bf16(a1, b0, acc[1][0]);
    acc[1][1] = wmma_bf16(a1, b1, acc[1][1]);
  }

#pragma unroll
  for (int im = 0; im < 2; ++im) {
#pragma unroll
    for (int ip = 0; ip < 2; ++ip) {
      const int mrow = mBase + im * 16 + g * 8;
      const int pcol = pBase + ip * 16 + s;
      v8f a = acc[im][ip];
      if (bias) {
#pragma unroll
        for (int r = 0; r < 8; ++r) a[r] += bias[mrow + r];
      }
      if (cbias) {
        const float* cb = cbias + (size_t)(pcol / cdiv) * M + mrow;
#pragma unroll
        for (int r = 0; r < 8; ++r) a[r] += cb[r];
      }
      if (relu) {
#pragma unroll
        for (int r = 0; r < 8; ++r) a[r] = fmaxf(a[r], 0.0f);
      }
      const size_t base = (size_t)pcol * M + mrow;
      if (Y) {
        PackU o;
#pragma unroll
        for (int r = 0; r < 8; ++r) o.h[r] = (bf16_t)a[r];
        *(uint4*)(Y + base) = o.q;
      }
      if (Yf) {
        *(float4*)(Yf + base)     = make_float4(a[0], a[1], a[2], a[3]);
        *(float4*)(Yf + base + 4) = make_float4(a[4], a[5], a[6], a[7]);
      }
    }
  }
}

// --------------------- small support kernels -------------------------------

// f32 -> bf16 with row/col zero-padding and optional column slice.
__global__ void k_cvt_pad(const float* __restrict__ src, bf16_t* __restrict__ dst,
                          int src_rows, int src_cols,
                          int dst_rows, int dst_cols, int col0)
{
  int i = blockIdx.x * blockDim.x + threadIdx.x;
  if (i >= dst_rows * dst_cols) return;
  int r = i / dst_cols, c = i % dst_cols;
  float v = 0.0f;
  int sc = c + col0;
  if (r < src_rows && sc < src_cols) v = src[(size_t)r * src_cols + sc];
  dst[i] = (bf16_t)v;
}

__global__ void k_pad_f32(const float* __restrict__ src, float* __restrict__ dst,
                          int n_src, int n_dst)
{
  int i = blockIdx.x * blockDim.x + threadIdx.x;
  if (i >= n_dst) return;
  dst[i] = (i < n_src) ? src[i] : 0.0f;
}

// input [B,3,N] f32 -> X0 [B*N, 32] bf16 (channels 3..31 zero)
__global__ void k_prep_input(const float* __restrict__ in, bf16_t* __restrict__ X0)
{
  int i = blockIdx.x * blockDim.x + threadIdx.x;      // over 16384*32
  if (i >= 16384 * 32) return;
  int p = i >> 5, c = i & 31;
  int b = p >> 11, n = p & 2047;
  float v = (c < 3) ? in[((size_t)b * 3 + c) * 2048 + n] : 0.0f;
  X0[i] = (bf16_t)v;
}

// channel-wise max over N points per batch; optional f32 and padded bf16 out.
__global__ void k_maxpool(const bf16_t* __restrict__ A, // [B*N, C]
                          float* __restrict__ gout,     // [B, C] f32 or null
                          bf16_t* __restrict__ gbf,     // [padRows, C] bf16 or null
                          int Bn, int Nn, int C, int padRows)
{
  int i = blockIdx.x * blockDim.x + threadIdx.x;       // over padRows*C
  if (i >= padRows * C) return;
  int row = i / C, c = i % C;
  float m = 0.0f;
  if (row < Bn) {
    const bf16_t* p = A + (size_t)row * Nn * C + c;
    m = -3.0e38f;
    for (int n = 0; n < Nn; ++n) m = fmaxf(m, (float)p[(size_t)n * C]);
  }
  if (gout && row < Bn) gout[(size_t)row * C + c] = m;
  if (gbf) gbf[i] = (bf16_t)m;
}

// per-batch column bias for e1 layer 0: vg[b,c] = e1_b0[c] + W[c,256:512].g[b]
__global__ void k_vg(const float* __restrict__ e1w0, const float* __restrict__ e1b0,
                     const float* __restrict__ g, float* __restrict__ vg)
{
  int i = blockIdx.x * blockDim.x + threadIdx.x;       // 8*512
  if (i >= 8 * 512) return;
  int b = i / 512, c = i % 512;
  const float* w  = e1w0 + (size_t)c * 512 + 256;
  const float* gb = g + (size_t)b * 256;
  float sum = e1b0[c];
  for (int k = 0; k < 256; ++k) sum += w[k] * gb[k];
  vg[i] = sum;
}

// Fused folding layer 0: h0[p,c] = relu( Tglob[b,c](incl. f_b0)
//        + fw0[c,0]*gx + fw0[c,1]*gy + fw0[c,2..4].coarse[b,k] )
__global__ void k_fold0(const float* __restrict__ Tg,     // [32, 512] f32
                        const float* __restrict__ coarse, // [32, 3072] f32
                        const float* __restrict__ fw0,    // [512, 1029] f32
                        bf16_t* __restrict__ H0,          // [npts, 512] bf16
                        int p0)
{
  int t = blockIdx.x * blockDim.x + threadIdx.x;        // npts * 64 threads
  int pp = t >> 6;
  int c0 = (t & 63) << 3;
  int p  = p0 + pp;
  if (p >= 131072) return;
  int b    = p >> 14;
  int nf   = p & 16383;
  int k    = nf >> 4;
  int gidx = nf & 15;
  const float L[4] = {-0.05f, -0.05f / 3.0f, 0.05f / 3.0f, 0.05f};
  float gx = L[gidx & 3], gy = L[gidx >> 2];
  const float* cp = coarse + (size_t)b * 3072 + k * 3;
  float px = cp[0], py = cp[1], pz = cp[2];
  const float* tg = Tg + (size_t)b * 512 + c0;
  PackU o;
#pragma unroll
  for (int r = 0; r < 8; ++r) {
    const float* w = fw0 + (size_t)(c0 + r) * 1029;
    float v = tg[r] + w[0] * gx + w[1] * gy + w[2] * px + w[3] * py + w[4] * pz;
    o.h[r] = (bf16_t)fmaxf(v, 0.0f);
  }
  *(uint4*)(H0 + (size_t)pp * 512 + c0) = o.q;
}

// fine = Yf2[:, 0:3] + repeated coarse;  writes d_out fine section.
__global__ void k_final(const float* __restrict__ Yf2,    // [npts, 32] f32
                        const float* __restrict__ coarse, // [32, 3072] f32
                        float* __restrict__ out, int p0, int np)
{
  int i = blockIdx.x * blockDim.x + threadIdx.x;         // np*3
  if (i >= np * 3) return;
  int pp = i / 3, j = i - pp * 3;
  int p = p0 + pp;
  int b = p >> 14;
  int k = (p & 16383) >> 4;
  out[24576 + (size_t)p * 3 + j] = Yf2[(size_t)pp * 32 + j]
                                 + coarse[(size_t)b * 3072 + k * 3 + j];
}

__global__ void k_copy_f32(const float* __restrict__ src, float* __restrict__ dst, int n)
{
  int i = blockIdx.x * blockDim.x + threadIdx.x;
  if (i < n) dst[i] = src[i];
}

// ---------------------------------------------------------------------------

static inline dim3 gemm_grid(int M, int P)     { return dim3((P + 63) / 64,  (M + 127) / 128); }
static inline dim3 gemm_grid_big(int M, int P) { return dim3((P + 255) / 256, (M + 127) / 128); }

extern "C" void kernel_launch(void* const* d_in, const int* in_sizes, int n_in,
                              void* d_out, int out_size, void* d_ws, size_t ws_size,
                              hipStream_t stream)
{
  (void)in_sizes; (void)n_in; (void)out_size; (void)ws_size;
  const float* input = (const float*)d_in[0];
  const float* e0_w0 = (const float*)d_in[1];  const float* e0_b0 = (const float*)d_in[2];
  const float* e0_w1 = (const float*)d_in[3];  const float* e0_b1 = (const float*)d_in[4];
  const float* e1_w0 = (const float*)d_in[5];  const float* e1_b0 = (const float*)d_in[6];
  const float* e1_w1 = (const float*)d_in[7];  const float* e1_b1 = (const float*)d_in[8];
  const float* d_w0  = (const float*)d_in[9];  const float* d_b0  = (const float*)d_in[10];
  const float* d_w1  = (const float*)d_in[11]; const float* d_b1  = (const float*)d_in[12];
  const float* d_w2  = (const float*)d_in[13]; const float* d_b2  = (const float*)d_in[14];
  const float* f_w0  = (const float*)d_in[15]; const float* f_b0  = (const float*)d_in[16];
  const float* f_w1  = (const float*)d_in[17]; const float* f_b1  = (const float*)d_in[18];
  const float* f_w2  = (const float*)d_in[19]; const float* f_b2  = (const float*)d_in[20];
  float* out = (float*)d_out;

  // ---- workspace layout (256B aligned, ~78 MB total) ----
  char* ws = (char*)d_ws;
  size_t off = 0;
  auto alloc = [&](size_t bytes) -> char* {
    char* p = ws + off;
    off = (off + bytes + 255) & ~(size_t)255;
    return p;
  };
  bf16_t* Wb_e0w0 = (bf16_t*)alloc(128  * 32   * 2);
  bf16_t* Wb_e0w1 = (bf16_t*)alloc(256  * 128  * 2);
  bf16_t* Wb_e1w0 = (bf16_t*)alloc(512  * 512  * 2);
  bf16_t* Wb_e1w1 = (bf16_t*)alloc(1024 * 512  * 2);
  bf16_t* Wb_dw0  = (bf16_t*)alloc(1024 * 1024 * 2);
  bf16_t* Wb_dw1  = (bf16_t*)alloc(1024 * 1024 * 2);
  bf16_t* Wb_dw2  = (bf16_t*)alloc(3072 * 1024 * 2);
  bf16_t* Wb_fw0g = (bf16_t*)alloc(512  * 1024 * 2);   // f_w0[:, 5:1029] packed
  bf16_t* Wb_fw1  = (bf16_t*)alloc(512  * 512  * 2);
  bf16_t* Wb_fw2  = (bf16_t*)alloc(32   * 512  * 2);   // f_w2 padded 3->32 rows
  float*  fb2p    = (float*) alloc(32 * 4);
  float*  g1      = (float*) alloc(8  * 256  * 4);
  float*  vg      = (float*) alloc(8  * 512  * 4);
  bf16_t* featbf  = (bf16_t*)alloc(32 * 1024 * 2);     // features padded to 32 rows
  bf16_t* Ad1     = (bf16_t*)alloc(32 * 1024 * 2);
  bf16_t* Ad2     = (bf16_t*)alloc(32 * 1024 * 2);
  float*  coarse  = (float*) alloc(32 * 3072 * 4);
  float*  Tg      = (float*) alloc(32 * 512  * 4);
  // encoder activations (64 MB block, overlaid by folding chunk buffers later)
  bf16_t* X0 = (bf16_t*)alloc((size_t)16384 * 32   * 2);
  bf16_t* A1 = (bf16_t*)alloc((size_t)16384 * 128  * 2);
  bf16_t* A2 = (bf16_t*)alloc((size_t)16384 * 256  * 2);
  bf16_t* A3 = (bf16_t*)alloc((size_t)16384 * 512  * 2);
  bf16_t* A4 = (bf16_t*)alloc((size_t)16384 * 1024 * 2);
  // folding per-chunk buffers overlay dead encoder buffers:
  bf16_t* H0  = (bf16_t*)X0;                                        // 16.8 MB
  bf16_t* H1  = (bf16_t*)A4;                                        // 16.8 MB
  float*  Yf2 = (float*)((char*)A4 + (size_t)16384 * 512 * 2);      // 2.1 MB

  const int T = 256;
  auto blocks = [](long n) { return (int)((n + 255) / 256); };

  // ---- weight conversion / padding (f32 -> bf16) ----
  k_cvt_pad<<<blocks(128 * 32),    T, 0, stream>>>(e0_w0, Wb_e0w0, 128, 3,    128, 32,   0);
  k_cvt_pad<<<blocks(256 * 128),   T, 0, stream>>>(e0_w1, Wb_e0w1, 256, 128,  256, 128,  0);
  k_cvt_pad<<<blocks(512 * 512),   T, 0, stream>>>(e1_w0, Wb_e1w0, 512, 512,  512, 512,  0);
  k_cvt_pad<<<blocks(1024 * 512),  T, 0, stream>>>(e1_w1, Wb_e1w1, 1024, 512, 1024, 512, 0);
  k_cvt_pad<<<blocks(1024 * 1024), T, 0, stream>>>(d_w0,  Wb_dw0,  1024, 1024, 1024, 1024, 0);
  k_cvt_pad<<<blocks(1024 * 1024), T, 0, stream>>>(d_w1,  Wb_dw1,  1024, 1024, 1024, 1024, 0);
  k_cvt_pad<<<blocks(3072 * 1024), T, 0, stream>>>(d_w2,  Wb_dw2,  3072, 1024, 3072, 1024, 0);
  k_cvt_pad<<<blocks(512 * 1024),  T, 0, stream>>>(f_w0,  Wb_fw0g, 512, 1029, 512, 1024, 5);
  k_cvt_pad<<<blocks(512 * 512),   T, 0, stream>>>(f_w1,  Wb_fw1,  512, 512,  512, 512,  0);
  k_cvt_pad<<<blocks(32 * 512),    T, 0, stream>>>(f_w2,  Wb_fw2,  3,   512,  32,  512,  0);
  k_pad_f32<<<1, 32, 0, stream>>>(f_b2, fb2p, 3, 32);

  // ---- encoder (heavy GEMMs: 64x64 wave tiles) ----
  k_prep_input<<<blocks(16384 * 32), T, 0, stream>>>(input, X0);
  k_gemm_big<<<gemm_grid_big(128, 16384), T, 0, stream>>>(Wb_e0w0, 32, X0, 32,
      e0_b0, nullptr, 1, A1, 128, 32, 16384, 1);
  k_gemm_big<<<gemm_grid_big(256, 16384), T, 0, stream>>>(Wb_e0w1, 128, A1, 128,
      e0_b1, nullptr, 1, A2, 256, 128, 16384, 0);
  k_maxpool<<<blocks(8 * 256), T, 0, stream>>>(A2, g1, nullptr, 8, 2048, 256, 8);
  k_vg<<<blocks(8 * 512), T, 0, stream>>>(e1_w0, e1_b0, g1, vg);
  // e1 layer 0: left 256 columns as WMMA GEMM, broadcast half folded into vg.
  k_gemm_big<<<gemm_grid_big(512, 16384), T, 0, stream>>>(Wb_e1w0, 512, A2, 256,
      nullptr, vg, 2048, A3, 512, 256, 16384, 1);
  k_gemm_big<<<gemm_grid_big(1024, 16384), T, 0, stream>>>(Wb_e1w1, 512, A3, 512,
      e1_b1, nullptr, 1, A4, 1024, 512, 16384, 0);
  k_maxpool<<<blocks(32 * 1024), T, 0, stream>>>(A4, nullptr, featbf, 8, 2048, 1024, 32);

  // ---- decoder (P padded to 32 columns; garbage columns never read) ----
  k_gemm<<<gemm_grid(1024, 32), T, 0, stream>>>(Wb_dw0, 1024, featbf, 1024,
      d_b0, nullptr, 1, Ad1, nullptr, 1024, 1024, 32, 1);
  k_gemm<<<gemm_grid(1024, 32), T, 0, stream>>>(Wb_dw1, 1024, Ad1, 1024,
      d_b1, nullptr, 1, Ad2, nullptr, 1024, 1024, 32, 1);
  k_gemm<<<gemm_grid(3072, 32), T, 0, stream>>>(Wb_dw2, 1024, Ad2, 1024,
      d_b2, nullptr, 1, nullptr, coarse, 3072, 1024, 32, 0);
  k_copy_f32<<<blocks(24576), T, 0, stream>>>(coarse, out, 24576);

  // ---- folding: T_global once per batch (includes f_b0) ----
  k_gemm<<<gemm_grid(512, 32), T, 0, stream>>>(Wb_fw0g, 1024, featbf, 1024,
      f_b0, nullptr, 1, nullptr, Tg, 512, 1024, 32, 0);

  // ---- folding pipeline in 8 chunks of 16384 points (L2-resident) ----
  const int CH = 16384;
  for (int ch = 0; ch < 8; ++ch) {
    int p0 = ch * CH;
    k_fold0<<<blocks((long)CH * 64), T, 0, stream>>>(Tg, coarse, f_w0, H0, p0);
    k_gemm_big<<<gemm_grid_big(512, CH), T, 0, stream>>>(Wb_fw1, 512, H0, 512,
        f_b1, nullptr, 1, H1, 512, 512, CH, 1);
    k_gemm<<<gemm_grid(32, CH), T, 0, stream>>>(Wb_fw2, 512, H1, 512,
        fb2p, nullptr, 1, nullptr, Yf2, 32, 512, CH, 0);
    k_final<<<blocks((long)CH * 3), T, 0, stream>>>(Yf2, coarse, out, p0, CH);
  }
}
